// Model_66700842107055
// MI455X (gfx1250) — compile-verified, matrix operand order checked by reading, not run on hardware
//
#include <hip/hip_runtime.h>
#include <hip/hip_bf16.h>
#include <math.h>

#define NATOM 2048
#define NTOT  4096          // B*N
#define DPAD  288           // 282 padded to multiple of 32
#define DREAL 282
#define EPSR  1e-16f

typedef __attribute__((ext_vector_type(16))) _Float16 v16h;
typedef __attribute__((ext_vector_type(8)))  _Float16 v8h;
typedef __attribute__((ext_vector_type(8)))  float    v8f;

__device__ __forceinline__ float wrapd(float d, float box) {
  return d - box * rintf(d / box);   // round-half-even like jnp.round
}

// ---------------------------------------------------------------------------
// Weight packing: stack both type-networks into one wide network.
//   G1 (128x288): layer0, Wb[n][k] (output-major, reduce dim contiguous)
//   G2 (128x128): layer1 block-diag      G3 (16x128): layer2 (col0=type0, col1=type1)
//   G4 (128x32), G5 (128x128), G6 (288x128): transposes for the backward GEMMs
// ---------------------------------------------------------------------------
__global__ void prep_weights(
    const float* __restrict__ W00, const float* __restrict__ b00,
    const float* __restrict__ W01, const float* __restrict__ b01,
    const float* __restrict__ W02, const float* __restrict__ b02,
    const float* __restrict__ W10, const float* __restrict__ b10,
    const float* __restrict__ W11, const float* __restrict__ b11,
    const float* __restrict__ W12, const float* __restrict__ b12,
    _Float16* __restrict__ G1, _Float16* __restrict__ G6,
    _Float16* __restrict__ G2, _Float16* __restrict__ G5,
    _Float16* __restrict__ G3, _Float16* __restrict__ G4,
    float* __restrict__ b1s, float* __restrict__ b2s, float* __restrict__ b3s)
{
  int idx = blockIdx.x * blockDim.x + threadIdx.x;
  if (idx < 128 * 288) {                       // G1[n][k] = W0stack[k][n]
    int n = idx / 288, k = idx % 288;
    float v = 0.f;
    if (k < DREAL) v = (n < 64) ? W00[k * 64 + n] : W10[k * 64 + (n - 64)];
    G1[idx] = (_Float16)v; return;
  }
  idx -= 128 * 288;
  if (idx < 288 * 128) {                       // G6 = G1^T  (288x128)
    int n = idx / 128, k = idx % 128;
    float v = 0.f;
    if (n < DREAL) v = (k < 64) ? W00[n * 64 + k] : W10[n * 64 + (k - 64)];
    G6[idx] = (_Float16)v; return;
  }
  idx -= 288 * 128;
  if (idx < 128 * 128) {                       // G2[n][k] block-diag W1
    int n = idx / 128, k = idx % 128;
    float v = 0.f;
    if (n < 64) { if (k < 64) v = W01[k * 64 + n]; }
    else        { if (k >= 64) v = W11[(k - 64) * 64 + (n - 64)]; }
    G2[idx] = (_Float16)v; return;
  }
  idx -= 128 * 128;
  if (idx < 128 * 128) {                       // G5 = G2^T
    int n = idx / 128, k = idx % 128;
    float v = 0.f;
    if (k < 64) { if (n < 64) v = W01[n * 64 + k]; }
    else        { if (n >= 64) v = W11[(n - 64) * 64 + (k - 64)]; }
    G5[idx] = (_Float16)v; return;
  }
  idx -= 128 * 128;
  if (idx < 16 * 128) {                        // G3 (16x128)
    int n = idx / 128, k = idx % 128;
    float v = 0.f;
    if (n == 0 && k < 64)        v = W02[k];
    else if (n == 1 && k >= 64)  v = W12[k - 64];
    G3[idx] = (_Float16)v; return;
  }
  idx -= 16 * 128;
  if (idx < 128 * 32) {                        // G4 (128x32) = G3^T zero-padded K
    int n = idx / 32, k = idx % 32;
    float v = 0.f;
    if (k == 0 && n < 64)        v = W02[n];
    else if (k == 1 && n >= 64)  v = W12[n - 64];
    G4[idx] = (_Float16)v; return;
  }
  idx -= 128 * 32;
  if (idx < 128) { b1s[idx] = (idx < 64) ? b00[idx] : b10[idx - 64]; return; }
  idx -= 128;
  if (idx < 128) { b2s[idx] = (idx < 64) ? b01[idx] : b11[idx - 64]; return; }
  idx -= 128;
  if (idx < 16)  { b3s[idx] = (idx == 0) ? b02[0] : (idx == 1 ? b12[0] : 0.f); }
}

// ---------------------------------------------------------------------------
// Per-atom neighbor search + descriptor build.
// One workgroup per atom; all batch xyz cached in LDS; single in-LDS bitonic
// sort on key = (type<<63)|(dist_bits<<16)|idx gives BOTH per-type sorted
// neighbor lists in one pass (IEEE positive floats are monotone in bits).
// ---------------------------------------------------------------------------
__global__ void __launch_bounds__(256) build_desc(
    const float* __restrict__ xyz, const int* __restrict__ types,
    const int* __restrict__ boxP,
    _Float16* __restrict__ descH, int* __restrict__ metaIdx,
    float* __restrict__ metaF)
{
  __shared__ float sx[NATOM], sy[NATOM], sz[NATOM];
  __shared__ unsigned long long keys[NATOM];
  __shared__ unsigned char sty[NATOM];
  __shared__ float sA[9];
  __shared__ int scnt0;

  const int am = blockIdx.x;
  const int b = am >> 11, i = am & (NATOM - 1);
  const float box = (float)boxP[0];
  const float* xb = xyz + (size_t)b * NATOM * 3;
  const int*   tb = types + (size_t)b * NATOM;

  for (int j = threadIdx.x; j < NATOM; j += blockDim.x) {
    sx[j] = xb[3 * j + 0]; sy[j] = xb[3 * j + 1]; sz[j] = xb[3 * j + 2];
    sty[j] = (unsigned char)(tb[j] & 1);
  }
  __syncthreads();
  const float xi = sx[i], yi = sy[i], zi = sz[i];

  for (int j = threadIdx.x; j < NATOM; j += blockDim.x) {
    float dx = wrapd(xi - sx[j] + EPSR, box);
    float dy = wrapd(yi - sy[j] + EPSR, box);
    float dz = wrapd(zi - sz[j] + EPSR, box);
    float d = sqrtf(dx * dx + dy * dy + dz * dz);
    unsigned long long key;
    if (j == i) key = ~0ULL;  // self: sorts past everything
    else key = ((unsigned long long)sty[j] << 63)
             | ((unsigned long long)__float_as_uint(d) << 16)
             | (unsigned long long)j;
    keys[j] = key;
  }
  __syncthreads();

  // bitonic sort, ascending, 2048 keys / 256 threads
  for (int k = 2; k <= NATOM; k <<= 1) {
    for (int j2 = k >> 1; j2 > 0; j2 >>= 1) {
      for (int t = threadIdx.x; t < NATOM; t += blockDim.x) {
        int ixj = t ^ j2;
        if (ixj > t) {
          bool up = ((t & k) == 0);
          unsigned long long a = keys[t], c = keys[ixj];
          if ((a > c) == up) { keys[t] = c; keys[ixj] = a; }
        }
      }
      __syncthreads();
    }
  }

  if (threadIdx.x == 0) {   // binary search for the type0/type1 boundary
    int lo = 0, hi = NATOM;
    while (lo < hi) { int mid = (lo + hi) >> 1; if ((keys[mid] >> 63) == 0ULL) lo = mid + 1; else hi = mid; }
    scnt0 = lo;
  }
  __syncthreads();
  const int cnt0 = scnt0;

  _Float16* drow = descH + (size_t)am * DPAD;
  int*   mi = metaIdx + (size_t)am * 140;
  float* mf = metaF + (size_t)am * 12;

  if (threadIdx.x < 138) {                     // radial: 46 type0 + 92 type1
    int t = threadIdx.x;
    unsigned long long key = (t < 46) ? keys[t] : keys[cnt0 + (t - 46)];
    float d = __uint_as_float((unsigned int)(key >> 16));
    int j = (int)(key & 0xFFFFULL);
    drow[t] = (_Float16)(1.0f / (d + EPSR));
    mi[t] = j;
  }

  if (threadIdx.x == 0) {                      // local frame (elementwise min per ref)
    unsigned long long k00 = keys[0], k01 = keys[1], k10 = keys[cnt0], k11 = keys[cnt0 + 1];
    float d00 = __uint_as_float((unsigned)(k00 >> 16)), d01 = __uint_as_float((unsigned)(k01 >> 16));
    float d10 = __uint_as_float((unsigned)(k10 >> 16)), d11 = __uint_as_float((unsigned)(k11 >> 16));
    int lf0 = (d10 < d00) ? (int)(k10 & 0xFFFF) : (int)(k00 & 0xFFFF);
    int lf1 = (d11 < d01) ? (int)(k11 & 0xFFFF) : (int)(k01 & 0xFFFF);
    float ld0 = fminf(d00, d10), ld1 = fminf(d01, d11);
    float ax = wrapd(xi - sx[lf0] + EPSR, box);
    float ay = wrapd(yi - sy[lf0] + EPSR, box);
    float az = wrapd(zi - sz[lf0] + EPSR, box);
    float bx = wrapd(xi - sx[lf1] + EPSR, box);
    float by = wrapd(yi - sy[lf1] + EPSR, box);
    float bz = wrapd(zi - sz[lf1] + EPSR, box);
    float inv0 = 1.f / (ld0 + EPSR), inv1 = 1.f / (ld1 + EPSR);
    float r0x = ax * inv0, r0y = ay * inv0, r0z = az * inv0;
    float r1x = bx * inv1, r1y = by * inv1, r1z = bz * inv1;
    float c = r0x * r1x + r0y * r1y + r0z * r1z;
    float wx = r1x - c * r0x, wy = r1y - c * r0y, wz = r1z - c * r0z;
    float wn = sqrtf(wx * wx + wy * wy + wz * wz);
    float ux = r0y * r1z - r0z * r1y, uy = r0z * r1x - r0x * r1z, uz = r0x * r1y - r0y * r1x;
    float un = sqrtf(ux * ux + uy * uy + uz * uz);
    sA[0] = r0x; sA[1] = r0y; sA[2] = r0z;
    sA[3] = wx / wn; sA[4] = wy / wn; sA[5] = wz / wn;
    sA[6] = ux / un; sA[7] = uy / un; sA[8] = uz / un;
    #pragma unroll
    for (int q = 0; q < 9; ++q) mf[q] = sA[q];
    mf[9] = ld0; mf[10] = ld1; mf[11] = 0.f;
    mi[138] = lf0; mi[139] = lf1;
  }
  __syncthreads();

  if (threadIdx.x < 48) {                      // angular: 16 type0 + 32 type1
    int q = threadIdx.x;
    int j = (q < 16) ? mi[q] : mi[46 + (q - 16)];
    float dx = wrapd(xi - sx[j] + EPSR, box);
    float dy = wrapd(yi - sy[j] + EPSR, box);
    float dz = wrapd(zi - sz[j] + EPSR, box);
    float d = sqrtf(dx * dx + dy * dy + dz * dz);
    float de = d + EPSR;
    float s = 1.f / (de * de);
    drow[138 + 3 * q + 0] = (_Float16)((sA[0] * dx + sA[1] * dy + sA[2] * dz) * s);
    drow[138 + 3 * q + 1] = (_Float16)((sA[3] * dx + sA[4] * dy + sA[5] * dz) * s);
    drow[138 + 3 * q + 2] = (_Float16)((sA[6] * dx + sA[7] * dy + sA[8] * dz) * s);
  } else if (threadIdx.x < 48 + (DPAD - DREAL)) {
    drow[DREAL + (threadIdx.x - 48)] = (_Float16)0.f;  // zero pad 282..287
  }
}

// ---------------------------------------------------------------------------
// WMMA GEMM: C[M,N] = act(A[M,K] * Wb[N,K]^T + bias)
// A,Wb f16 row-major; one 16x16 tile per wave; f32 accumulate via
// v_wmma_f32_16x16x32_f16. Fragment layout per CDNA5 ISA 7.12.2:
//   lanes 0-15: K 0-7 (v0-3), 16-23 (v4-7); lanes 16-31: K 8-15, 24-31.
// act: 0=none, 1=tanh, 2=multiply by (1-Ysave^2) (tanh backward)
// ---------------------------------------------------------------------------
__global__ void __launch_bounds__(256) gemm_wmma(
    const _Float16* __restrict__ A, int lda,
    const _Float16* __restrict__ Wb, int ldb,
    const float* __restrict__ bias, const float* __restrict__ Ysave,
    float* __restrict__ outF, _Float16* __restrict__ outH,
    int M, int N, int K, int act)
{
  const int lane = threadIdx.x & 31;
  const int wave = threadIdx.x >> 5;
  const int tilesN = N >> 4;
  const int tile = blockIdx.x * (blockDim.x >> 5) + wave;
  if (tile >= (M >> 4) * tilesN) return;       // wave-uniform: EXEC all-ones inside
  const int m0 = (tile / tilesN) << 4;
  const int n0 = (tile % tilesN) << 4;
  const int l15 = lane & 15, hi = lane >> 4;

  v8f acc = {0.f, 0.f, 0.f, 0.f, 0.f, 0.f, 0.f, 0.f};
  const _Float16* pa = A  + (size_t)(m0 + l15) * lda + hi * 8;
  const _Float16* pb = Wb + (size_t)(n0 + l15) * ldb + hi * 8;

  for (int k0 = 0; k0 < K; k0 += 32) {
    v8h alo = *(const v8h*)(pa + k0);
    v8h ahi = *(const v8h*)(pa + k0 + 16);
    v8h blo = *(const v8h*)(pb + k0);
    v8h bhi = *(const v8h*)(pb + k0 + 16);
    v16h av = __builtin_shufflevector(alo, ahi, 0,1,2,3,4,5,6,7,8,9,10,11,12,13,14,15);
    v16h bv = __builtin_shufflevector(blo, bhi, 0,1,2,3,4,5,6,7,8,9,10,11,12,13,14,15);
    acc = __builtin_amdgcn_wmma_f32_16x16x32_f16(false, av, false, bv,
                                                 (short)0, acc, false, false);
  }

  const int n = n0 + l15;
  const float bi = bias ? bias[n] : 0.f;
  #pragma unroll
  for (int r = 0; r < 8; ++r) {
    const int m = m0 + r + hi * 8;             // C/D layout: vgpr r, lane -> (m,n)
    float v = acc[r] + bi;
    if (act == 1) v = tanhf(v);
    else if (act == 2) { float y = Ysave[(size_t)m * N + n]; v = acc[r] * (1.f - y * y); }
    if (outF) outF[(size_t)m * N + n] = v;
    if (outH) outH[(size_t)m * N + n] = (_Float16)v;
  }
}

// ---------------------------------------------------------------------------
__global__ void finalize_e(const float* __restrict__ Z3, const int* __restrict__ types,
                           float* __restrict__ eAtom, _Float16* __restrict__ dZ3h)
{
  int m = blockIdx.x * blockDim.x + threadIdx.x;
  if (m >= NTOT) return;
  int ty = types[m] & 1;
  eAtom[m] = Z3[(size_t)m * 16 + ty];
  _Float16* row = dZ3h + (size_t)m * 32;
  #pragma unroll
  for (int c = 0; c < 32; ++c) row[c] = (_Float16)0.f;
  row[ty] = (_Float16)1.f;                     // dE/dZ3 seed (type select)
}

__global__ void reduce_energy(const float* __restrict__ eAtom, float* __restrict__ out)
{
  __shared__ float s[256];
  int b = blockIdx.x;
  float acc = 0.f;
  for (int i = threadIdx.x; i < NATOM; i += 256) acc += eAtom[b * NATOM + i];
  s[threadIdx.x] = acc; __syncthreads();
  for (int st = 128; st > 0; st >>= 1) {
    if (threadIdx.x < st) s[threadIdx.x] += s[threadIdx.x + st];
    __syncthreads();
  }
  if (threadIdx.x == 0) out[b] = s[0];
}

__global__ void zero_f32(float* __restrict__ p, int n)
{ int i = blockIdx.x * blockDim.x + threadIdx.x; if (i < n) p[i] = 0.f; }

__global__ void neg_copy(const float* __restrict__ src, float* __restrict__ dst, int n)
{ int i = blockIdx.x * blockDim.x + threadIdx.x; if (i < n) dst[i] = -src[i]; }

// ---------------------------------------------------------------------------
// Descriptor Jacobian: scatter force (= +dE/dx, negated later) contributions.
// threads 0..137: radial 1/(d+eps); 138..185: angular (through fixed A);
// thread 0: backprop through A (Gram-Schmidt + cross + normalizations).
// ---------------------------------------------------------------------------
__global__ void __launch_bounds__(192) desc_backward(
    const float* __restrict__ xyz, const int* __restrict__ boxP,
    const float* __restrict__ dDesc, const int* __restrict__ metaIdx,
    const float* __restrict__ metaF, float* __restrict__ gacc)
{
  __shared__ float sdA[9];
  __shared__ float sgxi[3];
  const int am = blockIdx.x;
  const int b = am >> 11, i = am & (NATOM - 1);
  const float box = (float)boxP[0];
  if (threadIdx.x < 9) sdA[threadIdx.x] = 0.f;
  if (threadIdx.x < 3) sgxi[threadIdx.x] = 0.f;
  __syncthreads();

  const float* xb = xyz + (size_t)b * NATOM * 3;
  const float xi = xb[3 * i], yi = xb[3 * i + 1], zi = xb[3 * i + 2];
  const int*   mi = metaIdx + (size_t)am * 140;
  const float* mf = metaF + (size_t)am * 12;
  const float* dd = dDesc + (size_t)am * DPAD;
  float* gb = gacc + (size_t)b * NATOM * 3;
  const int t = threadIdx.x;

  if (t < 138) {                               // radial
    int j = mi[t];
    float dx = wrapd(xi - xb[3 * j] + EPSR, box);
    float dy = wrapd(yi - xb[3 * j + 1] + EPSR, box);
    float dz = wrapd(zi - xb[3 * j + 2] + EPSR, box);
    float d = sqrtf(dx * dx + dy * dy + dz * dz);
    float de = d + EPSR;
    float coef = -dd[t] / (de * de * d);
    float gx = coef * dx, gy = coef * dy, gz = coef * dz;
    atomicAdd(&sgxi[0], gx); atomicAdd(&sgxi[1], gy); atomicAdd(&sgxi[2], gz);
    atomicAdd(&gb[3 * j + 0], -gx); atomicAdd(&gb[3 * j + 1], -gy); atomicAdd(&gb[3 * j + 2], -gz);
  } else if (t < 186) {                        // angular
    int q = t - 138;
    int j = (q < 16) ? mi[q] : mi[46 + (q - 16)];
    float dx = wrapd(xi - xb[3 * j] + EPSR, box);
    float dy = wrapd(yi - xb[3 * j + 1] + EPSR, box);
    float dz = wrapd(zi - xb[3 * j + 2] + EPSR, box);
    float d = sqrtf(dx * dx + dy * dy + dz * dz);
    float de = d + EPSR;
    float s = 1.f / (de * de);
    float g0 = dd[138 + 3 * q], g1 = dd[139 + 3 * q], g2 = dd[140 + 3 * q];
    float A00 = mf[0], A01 = mf[1], A02 = mf[2];
    float A10 = mf[3], A11 = mf[4], A12 = mf[5];
    float A20 = mf[6], A21 = mf[7], A22 = mf[8];
    float Ad0 = A00 * dx + A01 * dy + A02 * dz;
    float Ad1 = A10 * dx + A11 * dy + A12 * dz;
    float Ad2 = A20 * dx + A21 * dy + A22 * dz;
    float gAd = g0 * Ad0 + g1 * Ad1 + g2 * Ad2;
    float c2 = -2.f * gAd / (de * de * de * d);
    float gx = s * (A00 * g0 + A10 * g1 + A20 * g2) + c2 * dx;
    float gy = s * (A01 * g0 + A11 * g1 + A21 * g2) + c2 * dy;
    float gz = s * (A02 * g0 + A12 * g1 + A22 * g2) + c2 * dz;
    atomicAdd(&sgxi[0], gx); atomicAdd(&sgxi[1], gy); atomicAdd(&sgxi[2], gz);
    atomicAdd(&gb[3 * j + 0], -gx); atomicAdd(&gb[3 * j + 1], -gy); atomicAdd(&gb[3 * j + 2], -gz);
    atomicAdd(&sdA[0], g0 * dx * s); atomicAdd(&sdA[1], g0 * dy * s); atomicAdd(&sdA[2], g0 * dz * s);
    atomicAdd(&sdA[3], g1 * dx * s); atomicAdd(&sdA[4], g1 * dy * s); atomicAdd(&sdA[5], g1 * dz * s);
    atomicAdd(&sdA[6], g2 * dx * s); atomicAdd(&sdA[7], g2 * dy * s); atomicAdd(&sdA[8], g2 * dz * s);
  }
  __syncthreads();

  if (t == 0) {                                // frame backprop
    int lf0 = mi[138], lf1 = mi[139];
    float ax = wrapd(xi - xb[3 * lf0] + EPSR, box);
    float ay = wrapd(yi - xb[3 * lf0 + 1] + EPSR, box);
    float az = wrapd(zi - xb[3 * lf0 + 2] + EPSR, box);
    float bx = wrapd(xi - xb[3 * lf1] + EPSR, box);
    float by = wrapd(yi - xb[3 * lf1 + 1] + EPSR, box);
    float bz = wrapd(zi - xb[3 * lf1 + 2] + EPSR, box);
    float d0 = sqrtf(ax * ax + ay * ay + az * az);
    float d1 = sqrtf(bx * bx + by * by + bz * bz);
    float inv0 = 1.f / (d0 + EPSR), inv1 = 1.f / (d1 + EPSR);
    float r0x = ax * inv0, r0y = ay * inv0, r0z = az * inv0;
    float r1x = bx * inv1, r1y = by * inv1, r1z = bz * inv1;
    float c = r0x * r1x + r0y * r1y + r0z * r1z;
    float wx = r1x - c * r0x, wy = r1y - c * r0y, wz = r1z - c * r0z;
    float wn = sqrtf(wx * wx + wy * wy + wz * wz);
    float v2x = wx / wn, v2y = wy / wn, v2z = wz / wn;
    float ux = r0y * r1z - r0z * r1y, uy = r0z * r1x - r0x * r1z, uz = r0x * r1y - r0y * r1x;
    float un = sqrtf(ux * ux + uy * uy + uz * uz);
    float v3x = ux / un, v3y = uy / un, v3z = uz / un;
    float G0x = sdA[0], G0y = sdA[1], G0z = sdA[2];
    float G2x = sdA[3], G2y = sdA[4], G2z = sdA[5];
    float G3x = sdA[6], G3y = sdA[7], G3z = sdA[8];
    // v3 = u/|u|
    float gdv3 = G3x * v3x + G3y * v3y + G3z * v3z;
    float gux = (G3x - gdv3 * v3x) / un, guy = (G3y - gdv3 * v3y) / un, guz = (G3z - gdv3 * v3z) / un;
    // u = r0 x r1  ->  gr0 += r1 x gu ; gr1 += gu x r0
    float cr0x = r1y * guz - r1z * guy, cr0y = r1z * gux - r1x * guz, cr0z = r1x * guy - r1y * gux;
    float cr1x = guy * r0z - guz * r0y, cr1y = guz * r0x - gux * r0z, cr1z = gux * r0y - guy * r0x;
    // v2 = w/|w|
    float gdv2 = G2x * v2x + G2y * v2y + G2z * v2z;
    float gwx = (G2x - gdv2 * v2x) / wn, gwy = (G2y - gdv2 * v2y) / wn, gwz = (G2z - gdv2 * v2z) / wn;
    // w = r1 - c*r0 ;  c = r0.r1
    float gr1x = gwx + cr1x, gr1y = gwy + cr1y, gr1z = gwz + cr1z;
    float gr0x = -c * gwx + cr0x + G0x, gr0y = -c * gwy + cr0y + G0y, gr0z = -c * gwz + cr0z + G0z;
    float gc = -(gwx * r0x + gwy * r0y + gwz * r0z);
    gr0x += gc * r1x; gr0y += gc * r1y; gr0z += gc * r1z;
    gr1x += gc * r0x; gr1y += gc * r0y; gr1z += gc * r0z;
    // r0 = dm0/(d0+eps)
    float dp0 = gr0x * ax + gr0y * ay + gr0z * az;
    float t0 = inv0 * inv0 / d0;
    float gdm0x = inv0 * gr0x - dp0 * t0 * ax;
    float gdm0y = inv0 * gr0y - dp0 * t0 * ay;
    float gdm0z = inv0 * gr0z - dp0 * t0 * az;
    float dp1 = gr1x * bx + gr1y * by + gr1z * bz;
    float t1 = inv1 * inv1 / d1;
    float gdm1x = inv1 * gr1x - dp1 * t1 * bx;
    float gdm1y = inv1 * gr1y - dp1 * t1 * by;
    float gdm1z = inv1 * gr1z - dp1 * t1 * bz;
    atomicAdd(&gb[3 * lf0 + 0], -gdm0x); atomicAdd(&gb[3 * lf0 + 1], -gdm0y); atomicAdd(&gb[3 * lf0 + 2], -gdm0z);
    atomicAdd(&gb[3 * lf1 + 0], -gdm1x); atomicAdd(&gb[3 * lf1 + 1], -gdm1y); atomicAdd(&gb[3 * lf1 + 2], -gdm1z);
    atomicAdd(&gb[3 * i + 0], sgxi[0] + gdm0x + gdm1x);
    atomicAdd(&gb[3 * i + 1], sgxi[1] + gdm0y + gdm1y);
    atomicAdd(&gb[3 * i + 2], sgxi[2] + gdm0z + gdm1z);
  }
}

// ---------------------------------------------------------------------------
extern "C" void kernel_launch(void* const* d_in, const int* in_sizes, int n_in,
                              void* d_out, int out_size, void* d_ws, size_t ws_size,
                              hipStream_t stream) {
  (void)in_sizes; (void)n_in; (void)out_size; (void)ws_size;
  const float* xyz   = (const float*)d_in[0];
  const int*   types = (const int*)d_in[1];
  const int*   boxP  = (const int*)d_in[2];
  const float* W00 = (const float*)d_in[3],  *b00 = (const float*)d_in[4];
  const float* W01 = (const float*)d_in[5],  *b01 = (const float*)d_in[6];
  const float* W02 = (const float*)d_in[7],  *b02 = (const float*)d_in[8];
  const float* W10 = (const float*)d_in[9],  *b10 = (const float*)d_in[10];
  const float* W11 = (const float*)d_in[11], *b11 = (const float*)d_in[12];
  const float* W12 = (const float*)d_in[13], *b12 = (const float*)d_in[14];
  float* out = (float*)d_out;

  char* ws = (char*)d_ws;
  size_t off = 0;
  auto alloc = [&](size_t bytes) -> void* {
    void* p = ws + off;
    off = (off + bytes + 255) & ~(size_t)255;
    return p;
  };
  _Float16* descH = (_Float16*)alloc((size_t)NTOT * DPAD * 2);
  float*    Y1f   = (float*)alloc((size_t)NTOT * 128 * 4);
  _Float16* Y1h   = (_Float16*)alloc((size_t)NTOT * 128 * 2);
  float*    Y2f   = (float*)alloc((size_t)NTOT * 128 * 4);
  _Float16* Y2h   = (_Float16*)alloc((size_t)NTOT * 128 * 2);
  float*    Z3f   = (float*)alloc((size_t)NTOT * 16 * 4);
  _Float16* dZ3h  = (_Float16*)alloc((size_t)NTOT * 32 * 2);
  float*    dY2f  = (float*)alloc((size_t)NTOT * 128 * 4);
  _Float16* dY2h  = (_Float16*)alloc((size_t)NTOT * 128 * 2);
  float*    dY1f  = (float*)alloc((size_t)NTOT * 128 * 4);
  _Float16* dY1h  = (_Float16*)alloc((size_t)NTOT * 128 * 2);
  float*    dDesc = (float*)alloc((size_t)NTOT * DPAD * 4);
  _Float16* G1 = (_Float16*)alloc(128 * 288 * 2);
  _Float16* G6 = (_Float16*)alloc(288 * 128 * 2);
  _Float16* G2 = (_Float16*)alloc(128 * 128 * 2);
  _Float16* G5 = (_Float16*)alloc(128 * 128 * 2);
  _Float16* G3 = (_Float16*)alloc(16 * 128 * 2);
  _Float16* G4 = (_Float16*)alloc(128 * 32 * 2);
  float* b1s = (float*)alloc(128 * 4);
  float* b2s = (float*)alloc(128 * 4);
  float* b3s = (float*)alloc(16 * 4);
  int*   metaIdx = (int*)alloc((size_t)NTOT * 140 * 4);
  float* metaF   = (float*)alloc((size_t)NTOT * 12 * 4);
  float* eAtom   = (float*)alloc((size_t)NTOT * 4);
  float* gacc    = (float*)alloc((size_t)NTOT * 3 * 4);

  auto gemm = [&](const _Float16* A, int lda, const _Float16* Wb, int ldb,
                  const float* bias, const float* ys,
                  float* oF, _Float16* oH, int M, int N, int K, int act) {
    int tiles = (M / 16) * (N / 16);
    int blocks = (tiles + 7) / 8;
    gemm_wmma<<<blocks, 256, 0, stream>>>(A, lda, Wb, ldb, bias, ys, oF, oH, M, N, K, act);
  };

  // weight packing + zero force accumulator
  prep_weights<<<442, 256, 0, stream>>>(W00, b00, W01, b01, W02, b02,
                                        W10, b10, W11, b11, W12, b12,
                                        G1, G6, G2, G5, G3, G4, b1s, b2s, b3s);
  zero_f32<<<48, 256, 0, stream>>>(gacc, NTOT * 3);

  // descriptors
  build_desc<<<NTOT, 256, 0, stream>>>(xyz, types, boxP, descH, metaIdx, metaF);

  // forward MLP (both type-nets stacked to width 128)
  gemm(descH, DPAD, G1, DPAD, b1s, nullptr, Y1f, Y1h, NTOT, 128, DPAD, 1);
  gemm(Y1h, 128, G2, 128, b2s, nullptr, Y2f, Y2h, NTOT, 128, 128, 1);
  gemm(Y2h, 128, G3, 128, b3s, nullptr, Z3f, nullptr, NTOT, 16, 128, 0);

  // energy + backward seed
  finalize_e<<<(NTOT + 255) / 256, 256, 0, stream>>>(Z3f, types, eAtom, dZ3h);
  reduce_energy<<<2, 256, 0, stream>>>(eAtom, out);

  // backward MLP
  gemm(dZ3h, 32, G4, 32, nullptr, Y2f, dY2f, dY2h, NTOT, 128, 32, 2);
  gemm(dY2h, 128, G5, 128, nullptr, Y1f, dY1f, dY1h, NTOT, 128, 128, 2);
  gemm(dY1h, 128, G6, 128, nullptr, nullptr, dDesc, nullptr, NTOT, DPAD, 128, 0);

  // descriptor Jacobian -> forces
  desc_backward<<<NTOT, 192, 0, stream>>>(xyz, boxP, dDesc, metaIdx, metaF, gacc);
  neg_copy<<<48, 256, 0, stream>>>(gacc, out + 2, NTOT * 3);
}
